// MultiHeadSelfAttention_5171140624682
// MI455X (gfx1250) — matrix operand order repaired and checked by reading, not compiled, not run on hardware
//
#include <hip/hip_runtime.h>
#include <stdint.h>

#define S_LEN  4096
#define DMODEL 1024
#define NHEAD  16
#define DK     64

typedef __attribute__((ext_vector_type(16))) __bf16 v16bf;
typedef __attribute__((ext_vector_type(8)))  __bf16 v8bf;
typedef __attribute__((ext_vector_type(8)))  float  v8f;

static __device__ __forceinline__ unsigned short f2bf_u(float f) {
    union { float f; uint32_t u; } v; v.f = f;
    uint32_t u = v.u;
    uint32_t r = u + 0x7FFFu + ((u >> 16) & 1u);   // round-to-nearest-even
    return (unsigned short)(r >> 16);
}
static __device__ __forceinline__ float bf2f(unsigned short h) {
    union { uint32_t u; float f; } v; v.u = ((uint32_t)h) << 16;
    return v.f;
}
static __device__ __forceinline__ v16bf cat8(v8bf lo, v8bf hi) {
    v16bf r;
#pragma unroll
    for (int e = 0; e < 8; ++e) { r[e] = lo[e]; r[8 + e] = hi[e]; }
    return r;
}
static __device__ __forceinline__ float rmax16(float t) {
    t = fmaxf(t, __shfl_xor(t, 1, 32));
    t = fmaxf(t, __shfl_xor(t, 2, 32));
    t = fmaxf(t, __shfl_xor(t, 4, 32));
    t = fmaxf(t, __shfl_xor(t, 8, 32));
    return t;   // reduced within each 16-lane half
}
static __device__ __forceinline__ float rsum16(float t) {
    t += __shfl_xor(t, 1, 32);
    t += __shfl_xor(t, 2, 32);
    t += __shfl_xor(t, 4, 32);
    t += __shfl_xor(t, 8, 32);
    return t;
}

// ---------------- elementwise f32 -> bf16 ----------------
__global__ void f32_to_bf16_kernel(const float* __restrict__ in,
                                   unsigned short* __restrict__ out, int n) {
    int i = blockIdx.x * blockDim.x + threadIdx.x;
    if (i < n) out[i] = f2bf_u(in[i]);
}

// ---------------- bf16 NT GEMM: C[m,n] = sum_k A[m,k] * W[n,k] ----------------
// Block: 256 thr = 8 waves (2x4). Wave tile: 32(M) x 64(N). Block tile: 64 x 256.
template <bool OUT_F32>
__global__ __launch_bounds__(256) void gemm_bf16_nt_kernel(
    const unsigned short* __restrict__ A,   // [M][K] bf16 row-major
    const unsigned short* __restrict__ W,   // [N][K] bf16 row-major
    void* __restrict__ Cout,                // [M][N] (f32 or bf16)
    int M, int N, int K)
{
    const int lane  = threadIdx.x & 31;
    const int wave  = threadIdx.x >> 5;
    const int lm    = lane & 15;
    const int half  = lane >> 4;
    const int waveM = wave >> 2;   // 0..1
    const int waveN = wave & 3;    // 0..3
    const int mBase = blockIdx.x * 64 + waveM * 32;
    const int nBase = blockIdx.y * 256 + waveN * 64;

    v8f acc[2][4];
#pragma unroll
    for (int i = 0; i < 2; ++i)
#pragma unroll
        for (int j = 0; j < 4; ++j)
#pragma unroll
            for (int r = 0; r < 8; ++r) acc[i][j][r] = 0.0f;

    for (int k0 = 0; k0 < K; k0 += 32) {
        v16bf a[2];
#pragma unroll
        for (int i = 0; i < 2; ++i) {
            const unsigned short* pA = A + (size_t)(mBase + i * 16 + lm) * K + k0;
            v8bf lo = *(const v8bf*)(pA + half * 16);
            v8bf hi = *(const v8bf*)(pA + half * 16 + 8);
            a[i] = cat8(lo, hi);
        }
        v16bf b[4];
#pragma unroll
        for (int j = 0; j < 4; ++j) {
            const unsigned short* pB = W + (size_t)(nBase + j * 16 + lm) * K + k0 + half * 16;
            b[j] = *(const v16bf*)pB;
        }
#pragma unroll
        for (int i = 0; i < 2; ++i)
#pragma unroll
            for (int j = 0; j < 4; ++j)
                acc[i][j] = __builtin_amdgcn_wmma_f32_16x16x32_bf16(
                    false, a[i], false, b[j], (short)0, acc[i][j], false, false);
    }

#pragma unroll
    for (int i = 0; i < 2; ++i)
#pragma unroll
        for (int j = 0; j < 4; ++j)
#pragma unroll
            for (int r = 0; r < 8; ++r) {
                int row = mBase + i * 16 + r + 8 * half;
                int col = nBase + j * 16 + lm;
                if (OUT_F32)
                    ((float*)Cout)[(size_t)row * N + col] = acc[i][j][r];
                else
                    ((unsigned short*)Cout)[(size_t)row * N + col] = f2bf_u(acc[i][j][r]);
            }
}

// ---------------- RoPE in-place on bf16 [S][DMODEL] ----------------
__global__ void rope_kernel(unsigned short* __restrict__ X) {
    int idx = blockIdx.x * blockDim.x + threadIdx.x;     // over S*NHEAD*(DK/2)
    if (idx >= S_LEN * NHEAD * (DK / 2)) return;
    int i = idx & 31;            // rotary pair index
    int h = (idx >> 5) & 15;
    int s = idx >> 9;
    float freq = __expf(-((float)i * (1.0f / 32.0f)) * 9.210340371976184f); // 10000^(-i/32)
    float ang = (float)s * freq;
    float sn, cs;
    sincosf(ang, &sn, &cs);
    size_t off = (size_t)s * DMODEL + h * DK + 2 * i;
    float e = bf2f(X[off]), o = bf2f(X[off + 1]);
    X[off]     = f2bf_u(e * cs - o * sn);
    X[off + 1] = f2bf_u(e * sn + o * cs);
}

// ---------------- V transpose: Vb[S][DMODEL] -> Vt[DMODEL][S] ----------------
__global__ void transpose_v_kernel(const unsigned short* __restrict__ Vb,
                                   unsigned short* __restrict__ Vt) {
    int idx = blockIdx.x * blockDim.x + threadIdx.x;     // over S*DMODEL
    if (idx >= S_LEN * DMODEL) return;
    int c = idx & 1023;
    int s = idx >> 10;
    Vt[(size_t)c * S_LEN + s] = Vb[idx];
}

// ---------------- online-softmax step (C layout: lane holds col lm, rows r+8*half) ----
template <bool MASK>
static __device__ __forceinline__ void softmax_step(
    const v8f* sacc, float* mrow, float* lrow,
    float* corr, float* p0, float* p1,
    int q0, int key0, int lm, int half)
{
#pragma unroll
    for (int r = 0; r < 8; ++r) {
        const int rowg = q0 + r + 8 * half;
        float s0 = sacc[0][r] * 0.125f;
        float s1 = sacc[1][r] * 0.125f;
        if (MASK) {
            if (key0 + lm > rowg)      s0 = -1e30f;
            if (key0 + 16 + lm > rowg) s1 = -1e30f;
        }
        float rm = rmax16(fmaxf(s0, s1));
        float nm = fmaxf(mrow[r], rm);
        float cr = __expf(mrow[r] - nm);
        float e0 = __expf(s0 - nm);
        float e1 = __expf(s1 - nm);
        float rs = rsum16(e0 + e1);
        lrow[r] = lrow[r] * cr + rs;
        mrow[r] = nm;
        corr[r] = cr;
        p0[r] = e0; p1[r] = e1;
    }
}

// ---------------- Causal flash attention ----------------
// Block: 256 thr = 8 waves. Each wave owns one 16-query tile (8 adjacent tiles,
// same head). K (32x64) and V^T (64x32) blocks are staged into LDS with async
// b128 loads and shared by all 8 waves -> 8x less L2 traffic.
__global__ __launch_bounds__(256) void flash_attn_kernel(
    const unsigned short* __restrict__ Qb,   // [S][DMODEL] bf16 (post-RoPE)
    const unsigned short* __restrict__ Kb,   // [S][DMODEL] bf16 (post-RoPE)
    const unsigned short* __restrict__ Vt,   // [DMODEL][S] bf16
    unsigned short* __restrict__ Out)        // [S][DMODEL] bf16
{
    __shared__ __attribute__((aligned(128))) unsigned short kbuf[32 * 64];      // [key][d]
    __shared__ __attribute__((aligned(128))) unsigned short vbuf[64 * 32];      // [c][key]
    __shared__ __attribute__((aligned(128))) unsigned short pbuf[8][16 * 32];   // per-wave P

    const int t    = threadIdx.x;
    const int lane = t & 31;
    const int wave = t >> 5;
    const int lm   = lane & 15;
    const int half = lane >> 4;

    const int head  = blockIdx.x >> 5;           // 32 query-blocks per head
    const int qblk  = blockIdx.x & 31;           // 128 queries per block
    const int q0    = qblk * 128 + wave * 16;    // this wave's 16-query tile
    const int qmax0 = qblk * 128 + 7 * 16;       // last tile in block
    const int nkb   = (qmax0 + 47) >> 5;         // shared causal key-block count

    // staging addresses: one async b128 per thread per buffer (4KB each)
    const unsigned short* kBase = Kb + (size_t)head * DK;           // + key*1024 + d
    const unsigned short* vBase = Vt + (size_t)head * DK * S_LEN;   // + c*4096 + key
    const uint32_t ldsK = (uint32_t)(uintptr_t)(&kbuf[0]) + (uint32_t)t * 16u;
    const uint32_t ldsV = (uint32_t)(uintptr_t)(&vbuf[0]) + (uint32_t)t * 16u;
    const uint32_t ofK0 = ((uint32_t)(t >> 3) * 1024u + (uint32_t)(t & 7) * 8u) * 2u;
    const uint32_t ofV0 = ((uint32_t)(t >> 2) * 4096u + (uint32_t)(t & 3) * 8u) * 2u;

    // Q in A-layout, two d-steps (d 0..31, 32..63)
    v16bf qa[2];
    {
        const unsigned short* pQ = Qb + (size_t)(q0 + lm) * DMODEL + head * DK;
#pragma unroll
        for (int s = 0; s < 2; ++s) {
            v8bf lo = *(const v8bf*)(pQ + s * 32 + half * 16);
            v8bf hi = *(const v8bf*)(pQ + s * 32 + half * 16 + 8);
            qa[s] = cat8(lo, hi);
        }
    }

    float mrow[8], lrow[8];
    v8f o[4];
#pragma unroll
    for (int r = 0; r < 8; ++r) { mrow[r] = -1e30f; lrow[r] = 0.0f; }
#pragma unroll
    for (int ct = 0; ct < 4; ++ct)
#pragma unroll
        for (int r = 0; r < 8; ++r) o[ct][r] = 0.0f;

    unsigned short* pb = pbuf[wave];

    for (int kb = 0; kb < nkb; ++kb) {
        const int key0 = kb * 32;

        // ---- cooperative async staging of K and V blocks into LDS ----
        {
            const uint32_t ofK = ofK0 + (uint32_t)key0 * 2048u;
            const uint32_t ofV = ofV0 + (uint32_t)key0 * 2u;
            asm volatile("global_load_async_to_lds_b128 %0, %1, %2"
                         :: "v"(ldsK), "v"(ofK), "s"(kBase) : "memory");
            asm volatile("global_load_async_to_lds_b128 %0, %1, %2"
                         :: "v"(ldsV), "v"(ofV), "s"(vBase) : "memory");
            asm volatile("s_wait_asynccnt 0x0" ::: "memory");
        }
        __syncthreads();

        if (key0 <= q0 + 15) {   // wave-uniform: this block intersects the causal region
            // ---- S = Q K^T from LDS K block ----
            v8f sacc[2];
#pragma unroll
            for (int nt = 0; nt < 2; ++nt) {
                const unsigned short* pK = kbuf + (nt * 16 + lm) * 64;
                v16bf b0 = *(const v16bf*)(pK + half * 16);
                v16bf b1 = *(const v16bf*)(pK + 32 + half * 16);
                v8f z;
#pragma unroll
                for (int r = 0; r < 8; ++r) z[r] = 0.0f;
                z = __builtin_amdgcn_wmma_f32_16x16x32_bf16(false, qa[0], false, b0, (short)0, z, false, false);
                z = __builtin_amdgcn_wmma_f32_16x16x32_bf16(false, qa[1], false, b1, (short)0, z, false, false);
                sacc[nt] = z;
            }

            // ---- online softmax; mask only on diagonal blocks (wave-uniform) ----
            float p0[8], p1[8], corr[8];
            if (key0 + 31 > q0)
                softmax_step<true >(sacc, mrow, lrow, corr, p0, p1, q0, key0, lm, half);
            else
                softmax_step<false>(sacc, mrow, lrow, corr, p0, p1, q0, key0, lm, half);
#pragma unroll
            for (int ct = 0; ct < 4; ++ct)
#pragma unroll
                for (int r = 0; r < 8; ++r) o[ct][r] *= corr[r];

            // ---- P: C layout -> LDS row-major -> A layout ----
#pragma unroll
            for (int r = 0; r < 8; ++r) {
                const int row = r + 8 * half;
                pb[row * 32 + lm]      = f2bf_u(p0[r]);
                pb[row * 32 + 16 + lm] = f2bf_u(p1[r]);
            }
            asm volatile("s_wait_dscnt 0x0" ::: "memory");
            v16bf pa;
            {
                v8bf lo = *(const v8bf*)(pb + lm * 32 + half * 16);
                v8bf hi = *(const v8bf*)(pb + lm * 32 + half * 16 + 8);
                pa = cat8(lo, hi);
            }
            asm volatile("" ::: "memory");

            // ---- O += P * V from LDS V block (4 output column tiles) ----
#pragma unroll
            for (int ct = 0; ct < 4; ++ct) {
                const unsigned short* pV = vbuf + (ct * 16 + lm) * 32 + half * 16;
                v16bf vb = *(const v16bf*)pV;
                o[ct] = __builtin_amdgcn_wmma_f32_16x16x32_bf16(false, pa, false, vb, (short)0, o[ct], false, false);
            }
        }
        __syncthreads();   // protect LDS K/V from next iteration's staging
    }

    // ---- normalize and store ----
    float inv[8];
#pragma unroll
    for (int r = 0; r < 8; ++r) inv[r] = 1.0f / lrow[r];
#pragma unroll
    for (int ct = 0; ct < 4; ++ct)
#pragma unroll
        for (int r = 0; r < 8; ++r) {
            const int row = q0 + r + 8 * half;
            const int col = head * DK + ct * 16 + lm;
            Out[(size_t)row * DMODEL + col] = f2bf_u(o[ct][r] * inv[r]);
        }
}

// ---------------- host launcher ----------------
extern "C" void kernel_launch(void* const* d_in, const int* in_sizes, int n_in,
                              void* d_out, int out_size, void* d_ws, size_t ws_size,
                              hipStream_t stream) {
    (void)in_sizes; (void)n_in; (void)out_size; (void)ws_size;
    const float* x  = (const float*)d_in[0];
    const float* Wq = (const float*)d_in[1];
    const float* Wk = (const float*)d_in[2];
    const float* Wv = (const float*)d_in[3];
    const float* Wo = (const float*)d_in[4];

    char* ws = (char*)d_ws;
    unsigned short* xb   = (unsigned short*)(ws + (size_t)0);
    unsigned short* wqb  = (unsigned short*)(ws + ((size_t)8  << 20));
    unsigned short* wkb  = (unsigned short*)(ws + ((size_t)10 << 20));
    unsigned short* wvb  = (unsigned short*)(ws + ((size_t)12 << 20));
    unsigned short* wob  = (unsigned short*)(ws + ((size_t)14 << 20));
    unsigned short* Qb   = (unsigned short*)(ws + ((size_t)16 << 20));
    unsigned short* Kb   = (unsigned short*)(ws + ((size_t)24 << 20));
    unsigned short* Vb   = (unsigned short*)(ws + ((size_t)32 << 20));
    unsigned short* Vt   = (unsigned short*)(ws + ((size_t)40 << 20));
    unsigned short* attn = (unsigned short*)(ws + ((size_t)48 << 20));

    const int nx = S_LEN * DMODEL;       // 4,194,304
    const int nw = DMODEL * DMODEL;      // 1,048,576
    f32_to_bf16_kernel<<<(nx + 255) / 256, 256, 0, stream>>>(x,  xb,  nx);
    f32_to_bf16_kernel<<<(nw + 255) / 256, 256, 0, stream>>>(Wq, wqb, nw);
    f32_to_bf16_kernel<<<(nw + 255) / 256, 256, 0, stream>>>(Wk, wkb, nw);
    f32_to_bf16_kernel<<<(nw + 255) / 256, 256, 0, stream>>>(Wv, wvb, nw);
    f32_to_bf16_kernel<<<(nw + 255) / 256, 256, 0, stream>>>(Wo, wob, nw);

    dim3 gg(S_LEN / 64, DMODEL / 256), bb(256);
    gemm_bf16_nt_kernel<false><<<gg, bb, 0, stream>>>(xb, wqb, Qb, S_LEN, DMODEL, DMODEL);
    gemm_bf16_nt_kernel<false><<<gg, bb, 0, stream>>>(xb, wkb, Kb, S_LEN, DMODEL, DMODEL);
    gemm_bf16_nt_kernel<false><<<gg, bb, 0, stream>>>(xb, wvb, Vb, S_LEN, DMODEL, DMODEL);

    const int nr = S_LEN * NHEAD * (DK / 2);   // 2,097,152
    rope_kernel<<<(nr + 255) / 256, 256, 0, stream>>>(Qb);
    rope_kernel<<<(nr + 255) / 256, 256, 0, stream>>>(Kb);

    transpose_v_kernel<<<(nx + 255) / 256, 256, 0, stream>>>(Vb, Vt);

    flash_attn_kernel<<<NHEAD * (S_LEN / 128), 256, 0, stream>>>(Qb, Kb, Vt, attn);

    gemm_bf16_nt_kernel<true><<<gg, bb, 0, stream>>>(attn, wob, d_out, S_LEN, DMODEL, DMODEL);
}
